// DynamicTokenLevelCrossAttention_28896539968072
// MI455X (gfx1250) — matrix operand order repaired and checked by reading, not compiled, not run on hardware
//
#include <hip/hip_runtime.h>
#include <hip/hip_bf16.h>

// ---------------------------------------------------------------------------
// CDNA5 (gfx1250) fused cross-attention, wave32 WMMA f16->f32,
// with async global->LDS staging (ASYNCcnt) in the attention kernel.
// ---------------------------------------------------------------------------

typedef __attribute__((ext_vector_type(8)))  _Float16 v8h;
typedef __attribute__((ext_vector_type(16))) _Float16 v16h;
typedef __attribute__((ext_vector_type(8)))  float    v8f;

#define B_DIM  4
#define N_TOK  4096
#define D_DIM  768
#define K_DESC 64
#define DS_DIM 32

// D = A(16x32 f16) * B(32x16 f16) + C(16x16 f32)
__device__ __forceinline__ v8f wmma_f16(v16h a, v16h b, v8f c) {
    return __builtin_amdgcn_wmma_f32_16x16x32_f16(
        /*neg_a=*/false, a, /*neg_b=*/false, b,
        /*c_mod=*/(short)0, c, /*reuse_a=*/false, /*reuse_b=*/false);
}

// A fragment: row-major 16x32 tile, lda in elements.
// lanes 0-15 hold M=lane, K = {0..7, 16..23}; lanes 16-31: K = {8..15, 24..31}
__device__ __forceinline__ v16h load_a(const _Float16* p, int lda, int lane) {
    int row = lane & 15, hi = (lane >> 4) & 1;
    const _Float16* q = p + (size_t)row * lda + hi * 8;
    union { v16h v; v8h h[2]; } u;
    u.h[0] = *(const v8h*)(q);
    u.h[1] = *(const v8h*)(q + 16);
    return u.v;
}

// B fragment: B[Kdim][N] where each column N is CONTIGUOUS in memory
// (memory "rows" indexed by N with leading dim ldb along Kdim).
// lanes 0-15 hold N=lane, K = {0..7, 16..23}; lanes 16-31: K = {8..15, 24..31}
__device__ __forceinline__ v16h load_bT(const _Float16* p, int ldb, int lane) {
    int coln = lane & 15, hi = (lane >> 4) & 1;
    const _Float16* q = p + (size_t)coln * ldb + hi * 8;
    union { v16h v; v8h h[2]; } u;
    u.h[0] = *(const v8h*)(q);
    u.h[1] = *(const v8h*)(q + 16);
    return u.v;
}

// Async global->LDS 16-byte copy (per lane), tracked by ASYNCcnt.
__device__ __forceinline__ void async_ld_b128(unsigned lds_addr, const void* gaddr) {
    asm volatile("global_load_async_to_lds_b128 %0, %1, off"
                 :: "v"(lds_addr), "v"(gaddr)
                 : "memory");
}

// ---------------------------------------------------------------------------
// f32 -> f16 convert
// ---------------------------------------------------------------------------
__global__ void cvt_f16_kernel(const float* __restrict__ in,
                               _Float16* __restrict__ out, int n) {
    int i = blockIdx.x * blockDim.x + threadIdx.x;
    int stride = gridDim.x * blockDim.x;
    for (; i < n; i += stride) out[i] = (_Float16)in[i];
}

// ---------------------------------------------------------------------------
// Projections: k_note = x @ Wk^T (row-major f16), vT = (x @ Wv^T)^T
// Block: 256 thr (8 waves); block tile M=32 x N=128; wave tile M=16 x N=32.
// ---------------------------------------------------------------------------
__global__ __launch_bounds__(256) void proj_kernel(
    const _Float16* __restrict__ xh,
    const _Float16* __restrict__ wkh,
    const _Float16* __restrict__ wvh,
    _Float16* __restrict__ knote,   // [4][4096][768]
    _Float16* __restrict__ vT)      // [4][768][4096]
{
    const int tid = threadIdx.x, lane = tid & 31, w = tid >> 5;
    const int mi = w & 1, nseg = w >> 1;
    const int m0 = blockIdx.x * 32 + mi * 16;
    const int n0 = blockIdx.y * 128 + nseg * 32;

    v8f aK[2] = {v8f{}, v8f{}};
    v8f aV[2] = {v8f{}, v8f{}};
    for (int kk = 0; kk < D_DIM; kk += 32) {
        v16h a = load_a(xh + (size_t)m0 * D_DIM + kk, D_DIM, lane);
        // prefetch next A slab toward WGP$/L2
        __builtin_prefetch(xh + (size_t)m0 * D_DIM + kk + 64, 0, 1);
        for (int j = 0; j < 2; ++j) {
            v16h bk = load_bT(wkh + (size_t)(n0 + j * 16) * D_DIM + kk, D_DIM, lane);
            v16h bv = load_bT(wvh + (size_t)(n0 + j * 16) * D_DIM + kk, D_DIM, lane);
            aK[j] = wmma_f16(a, bk, aK[j]);
            aV[j] = wmma_f16(a, bv, aV[j]);
        }
    }
    const int col = lane & 15, rbase = ((lane >> 4) & 1) * 8;
    for (int j = 0; j < 2; ++j) {
        for (int r = 0; r < 8; ++r) {
            int row = m0 + rbase + r;        // global token row [0,16384)
            int o   = n0 + j * 16 + col;     // output feature
            knote[(size_t)row * D_DIM + o] = (_Float16)aK[j][r];
            int b = row >> 12, t = row & (N_TOK - 1);
            vT[((size_t)b * D_DIM + o) * N_TOK + t] = (_Float16)aV[j][r];
        }
    }
}

// ---------------------------------------------------------------------------
// Async-staged S-tile: S(32 x 16 tokens), K streamed through per-wave LDS
// double buffer in 16tok x 64feat stages (2 KB each, 4 b128 issues/lane-wave).
// ---------------------------------------------------------------------------
#define STAGE_F   64                   // features per stage
#define STAGE_ELT (16 * STAGE_F)       // 1024 f16 = 2048 B per buffer

// Issue one stage: 16 tokens x 64 features from global (row stride 768) -> LDS.
__device__ __forceinline__ void stage_issue(const _Float16* gbase,  // token0, feat0
                                            unsigned lds_base, int lane) {
    const int tok = lane & 15, hi = (lane >> 4) & 1;
    const char* g = (const char*)(gbase + (size_t)tok * D_DIM) + hi * 16;
    unsigned    l = lds_base + tok * (STAGE_F * 2) + hi * 16;
    async_ld_b128(l,      g);
    async_ld_b128(l + 32, g + 32);
    async_ld_b128(l + 64, g + 64);
    async_ld_b128(l + 96, g + 96);
}

// Accumulate full-K (768) S subtile for this wave's 16 tokens.
__device__ __forceinline__ void s_tile_async(
    const _Float16* __restrict__ A0,      // enc rows for this k (global)
    const _Float16* __restrict__ Kb_tok,  // k_note row of this wave's token0
    _Float16* kst,                        // shared: this wave's [2][16][64]
    int lane, v8f& acc0, v8f& acc1)
{
    const unsigned lbase = (unsigned)(uintptr_t)kst;   // LDS byte offset
    stage_issue(Kb_tok, lbase, lane);                  // stage 0 -> buf 0
    for (int s = 0; s < D_DIM / STAGE_F; ++s) {        // 12 stages
        if (s < D_DIM / STAGE_F - 1) {
            // buffer (s+1)&1 was last read two stages ago; make sure those
            // ds reads retired before async writes can land in it.
            asm volatile("s_wait_dscnt 0x0" ::: "memory");
            stage_issue(Kb_tok + (s + 1) * STAGE_F,
                        lbase + (unsigned)(((s + 1) & 1) * (STAGE_ELT * 2)), lane);
            asm volatile("s_wait_asynccnt 0x4" ::: "memory"); // stage s landed
        } else {
            asm volatile("s_wait_asynccnt 0x0" ::: "memory");
        }
        const _Float16* kb = kst + (s & 1) * STAGE_ELT;
        for (int kk = 0; kk < STAGE_F; kk += 32) {
            v16h a0 = load_a(A0 + s * STAGE_F + kk, D_DIM, lane);
            v16h a1 = load_a(A0 + 16 * D_DIM + s * STAGE_F + kk, D_DIM, lane);
            v16h bf = load_bT(kb + kk, STAGE_F, lane);   // from LDS
            acc0 = wmma_f16(a0, bf, acc0);
            acc1 = wmma_f16(a1, bf, acc1);
        }
    }
}

// ---------------------------------------------------------------------------
// Attention per (b,k): online softmax stats over n, then c[n] = sum_ds
// alpha[ds]*exp(s-m).  Block: 256 thr (8 waves), n chunks of 128.
// ---------------------------------------------------------------------------
__global__ __launch_bounds__(256) void attn_cvec_kernel(
    const _Float16* __restrict__ ench,   // [64][32][768]
    const _Float16* __restrict__ knote,  // [4][4096][768]
    const int* __restrict__ dmask,       // [64][32]
    const int* __restrict__ nmask,       // [4][4096]
    _Float16* __restrict__ Cw)           // [4][64][4096]
{
    const int k = blockIdx.x, b = blockIdx.y;
    const int tid = threadIdx.x, lane = tid & 31, w = tid >> 5;
    const float scale = 0.03608439182435161f;   // 1/sqrt(768)

    __shared__ _Float16 Kstage[8][2][STAGE_ELT];   // 32 KB, per-wave buffers
    __shared__ float Sbuf[DS_DIM * 128];           // 16 KB
    __shared__ float red[256];
    __shared__ float mrow[DS_DIM], lrow[DS_DIM], arow[DS_DIM];
    __shared__ int   dmk[DS_DIM];

    if (tid < DS_DIM) {
        dmk[tid]  = dmask[k * DS_DIM + tid];
        mrow[tid] = -3.0e38f;
        lrow[tid] = 0.0f;
    }
    __syncthreads();

    const _Float16* A0 = ench + (size_t)k * DS_DIM * D_DIM;
    const _Float16* Kb = knote + (size_t)b * N_TOK * D_DIM;
    _Float16* kst = &Kstage[w][0][0];
    const int col = lane & 15, rbase = ((lane >> 4) & 1) * 8;

    // ---- Pass 1: online row max / exp-sum ----
    for (int n0 = 0; n0 < N_TOK; n0 += 128) {
        const int t = n0 + w * 16 + col;
        v8f acc0 = v8f{}, acc1 = v8f{};
        s_tile_async(A0, Kb + (size_t)(n0 + w * 16) * D_DIM, kst, lane, acc0, acc1);

        const int cm = nmask[b * N_TOK + t];
        for (int r = 0; r < 8; ++r) {
            int r0 = rbase + r, r1 = 16 + rbase + r;
            float s0 = (cm && dmk[r0]) ? acc0[r] * scale : -1e9f;
            float s1 = (cm && dmk[r1]) ? acc1[r] * scale : -1e9f;
            Sbuf[r0 * 128 + w * 16 + col] = s0;
            Sbuf[r1 * 128 + w * 16 + col] = s1;
        }
        __syncthreads();
        {   // partial row max: 32 rows x 8 segments of 16
            int row = tid >> 3, seg = tid & 7;
            const float* p = &Sbuf[row * 128 + seg * 16];
            float mx = p[0];
            for (int i = 1; i < 16; ++i) mx = fmaxf(mx, p[i]);
            red[row * 8 + seg] = mx;
        }
        __syncthreads();
        if (tid < DS_DIM) {
            float mx = red[tid * 8];
            for (int i = 1; i < 8; ++i) mx = fmaxf(mx, red[tid * 8 + i]);
            float mnew = fmaxf(mrow[tid], mx);
            lrow[tid] *= __expf(mrow[tid] - mnew);
            mrow[tid] = mnew;
        }
        __syncthreads();
        {   // partial exp-sum with updated max
            int row = tid >> 3, seg = tid & 7;
            float mn = mrow[row];
            const float* p = &Sbuf[row * 128 + seg * 16];
            float sm = 0.0f;
            for (int i = 0; i < 16; ++i) sm += __expf(p[i] - mn);
            red[row * 8 + seg] = sm;
        }
        __syncthreads();
        if (tid < DS_DIM) {
            float sm = 0.0f;
            for (int i = 0; i < 8; ++i) sm += red[tid * 8 + i];
            lrow[tid] += sm;
        }
        __syncthreads();
    }

    // ---- desc_weights: softmax over ds of (mask ? 1 : -1e9) == uniform ----
    if (tid < DS_DIM) red[tid] = dmk[tid] ? 1.0f : 0.0f;
    __syncthreads();
    if (tid == 0) {
        float c = 0.0f;
        for (int i = 0; i < DS_DIM; ++i) c += red[i];
        red[DS_DIM] = (c > 0.0f) ? 1.0f / c : 0.0f;
    }
    __syncthreads();
    if (tid < DS_DIM) {
        float wgt = dmk[tid] ? red[DS_DIM] : 0.0f;
        float l   = lrow[tid];
        arow[tid] = (l > 0.0f) ? wgt / l : 0.0f;
    }
    __syncthreads();

    // ---- Pass 2: c[t] = sum_ds alpha[ds] * exp(s - m) ----
    for (int n0 = 0; n0 < N_TOK; n0 += 128) {
        const int t = n0 + w * 16 + col;
        v8f acc0 = v8f{}, acc1 = v8f{};
        s_tile_async(A0, Kb + (size_t)(n0 + w * 16) * D_DIM, kst, lane, acc0, acc1);

        const int cm = nmask[b * N_TOK + t];
        float cpart = 0.0f;
        for (int r = 0; r < 8; ++r) {
            int r0 = rbase + r, r1 = 16 + rbase + r;
            if (cm && dmk[r0]) cpart += __expf(acc0[r] * scale - mrow[r0]) * arow[r0];
            if (cm && dmk[r1]) cpart += __expf(acc1[r] * scale - mrow[r1]) * arow[r1];
        }
        // lanes L and L+16 hold different rows of the SAME token column
        cpart += __shfl_xor(cpart, 16, 32);
        if (lane < 16)
            Cw[((size_t)(b * K_DESC + k)) * N_TOK + t] = (_Float16)cpart;
    }
}

// ---------------------------------------------------------------------------
// pooled_b[64 x 768] = C_b[64 x 4096] @ V_b[4096 x 768]  (V stored transposed)
// Block: 256 thr; block tile M=64 x N=128; wave tile M=16 x N=64.
// ---------------------------------------------------------------------------
__global__ __launch_bounds__(256) void pooled_kernel(
    const _Float16* __restrict__ Cw,   // [4][64][4096]
    const _Float16* __restrict__ vT,   // [4][768][4096]
    float* __restrict__ pooled)        // [4][64][768]
{
    const int b = blockIdx.x;
    const int tid = threadIdx.x, lane = tid & 31, w = tid >> 5;
    const int mi = w >> 1;                       // 0..3 -> M tile of 16
    const int n0 = blockIdx.y * 128 + (w & 1) * 64;

    const _Float16* A0 = Cw + (size_t)b * K_DESC * N_TOK + (size_t)(mi * 16) * N_TOK;
    const _Float16* B0 = vT + (size_t)b * D_DIM * N_TOK;

    v8f acc[4] = {v8f{}, v8f{}, v8f{}, v8f{}};
    for (int t = 0; t < N_TOK; t += 32) {
        v16h a = load_a(A0 + t, N_TOK, lane);
        __builtin_prefetch(A0 + t + 64, 0, 1);
        for (int j = 0; j < 4; ++j) {
            v16h bf = load_bT(B0 + (size_t)(n0 + j * 16) * N_TOK + t, N_TOK, lane);
            acc[j] = wmma_f16(a, bf, acc[j]);
        }
    }
    const int col = lane & 15, rbase = ((lane >> 4) & 1) * 8;
    for (int j = 0; j < 4; ++j) {
        for (int r = 0; r < 8; ++r) {
            int row = mi * 16 + rbase + r;   // k index 0..63
            int o   = n0 + j * 16 + col;
            pooled[((size_t)b * K_DESC + row) * D_DIM + o] = acc[j][r];
        }
    }
}

// ---------------------------------------------------------------------------
// LayerNorm(d=768, eps=1e-5, population var) + dot(Wo) + bo -> out[b*64+k]
// ---------------------------------------------------------------------------
__global__ __launch_bounds__(256) void ln_out_kernel(
    const float* __restrict__ pooled,   // [256][768]
    const float* __restrict__ gamma,
    const float* __restrict__ beta,
    const float* __restrict__ Wo,       // [768]
    const float* __restrict__ bo,       // [1]
    float* __restrict__ out)            // [256]
{
    const int row = blockIdx.x, tid = threadIdx.x;
    __shared__ float sred[256];
    const float* p = pooled + (size_t)row * D_DIM;

    float v0 = p[tid], v1 = p[tid + 256], v2 = p[tid + 512];
    sred[tid] = v0 + v1 + v2;
    __syncthreads();
    for (int st = 128; st > 0; st >>= 1) {
        if (tid < st) sred[tid] += sred[tid + st];
        __syncthreads();
    }
    const float mean = sred[0] * (1.0f / D_DIM);
    __syncthreads();

    sred[tid] = v0 * v0 + v1 * v1 + v2 * v2;
    __syncthreads();
    for (int st = 128; st > 0; st >>= 1) {
        if (tid < st) sred[tid] += sred[tid + st];
        __syncthreads();
    }
    const float var  = sred[0] * (1.0f / D_DIM) - mean * mean;
    const float rstd = rsqrtf(var + 1e-5f);
    __syncthreads();

    float dotp = 0.0f;
    for (int i = tid; i < D_DIM; i += 256) {
        float y = (p[i] - mean) * rstd * gamma[i] + beta[i];
        dotp += y * Wo[i];
    }
    sred[tid] = dotp;
    __syncthreads();
    for (int st = 128; st > 0; st >>= 1) {
        if (tid < st) sred[tid] += sred[tid + st];
        __syncthreads();
    }
    if (tid == 0) out[row] = sred[0] + bo[0];
}

// ---------------------------------------------------------------------------
extern "C" void kernel_launch(void* const* d_in, const int* in_sizes, int n_in,
                              void* d_out, int out_size, void* d_ws, size_t ws_size,
                              hipStream_t stream) {
    (void)in_sizes; (void)n_in; (void)out_size; (void)ws_size;

    const float* x     = (const float*)d_in[0];   // [4,4096,768]
    const float* enc   = (const float*)d_in[1];   // [64,32,768]
    const int*   dmask = (const int*)  d_in[2];   // [64,32]
    const int*   nmask = (const int*)  d_in[3];   // [4,4096]
    const float* Wk    = (const float*)d_in[4];   // [768,768]
    const float* Wv    = (const float*)d_in[5];   // [768,768]
    const float* Wo    = (const float*)d_in[6];   // [1,768]
    const float* bo    = (const float*)d_in[7];   // [1]
    const float* gam   = (const float*)d_in[8];   // [768]
    const float* bet   = (const float*)d_in[9];   // [768]
    float* out = (float*)d_out;                   // [4,64] f32

    char* ws = (char*)d_ws;
    size_t off = 0;
    auto alloc = [&](size_t bytes) -> void* {
        void* p = ws + off;
        off += (bytes + 255) & ~(size_t)255;
        return p;
    };
    _Float16* xh     = (_Float16*)alloc((size_t)16384 * 768 * 2);
    _Float16* wkh    = (_Float16*)alloc((size_t)768 * 768 * 2);
    _Float16* wvh    = (_Float16*)alloc((size_t)768 * 768 * 2);
    _Float16* ench   = (_Float16*)alloc((size_t)64 * 32 * 768 * 2);
    _Float16* knote  = (_Float16*)alloc((size_t)16384 * 768 * 2);
    _Float16* vT     = (_Float16*)alloc((size_t)4 * 768 * 4096 * 2);
    _Float16* Cw     = (_Float16*)alloc((size_t)4 * 64 * 4096 * 2);
    float*    pooled = (float*)   alloc((size_t)4 * 64 * 768 * 4);

    cvt_f16_kernel<<<2048, 256, 0, stream>>>(x,   xh,   16384 * 768);
    cvt_f16_kernel<<<512,  256, 0, stream>>>(Wk,  wkh,  768 * 768);
    cvt_f16_kernel<<<512,  256, 0, stream>>>(Wv,  wvh,  768 * 768);
    cvt_f16_kernel<<<768,  256, 0, stream>>>(enc, ench, 64 * 32 * 768);

    proj_kernel     <<<dim3(512, 6), 256, 0, stream>>>(xh, wkh, wvh, knote, vT);
    attn_cvec_kernel<<<dim3(64, 4),  256, 0, stream>>>(ench, knote, dmask, nmask, Cw);
    pooled_kernel   <<<dim3(4, 6),   256, 0, stream>>>(Cw, vT, pooled);
    ln_out_kernel   <<<256, 256, 0, stream>>>(pooled, gam, bet, Wo, bo, out);
}
